// SelfAttention_65712999628905
// MI455X (gfx1250) — compile-verified, matrix operand order checked by reading, not run on hardware
//
#include <hip/hip_runtime.h>
#include <hip/hip_bf16.h>

// ---------------------------------------------------------------------------
// Types for CDNA5 WMMA (wave32): v_wmma_f32_16x16x32_bf16
// ---------------------------------------------------------------------------
typedef __attribute__((ext_vector_type(16))) __bf16 bf16x16;
typedef __attribute__((ext_vector_type(8)))  __bf16 bf16x8;
typedef __attribute__((ext_vector_type(8)))  float  f32x8;

static __device__ __forceinline__ bf16x16 cat8(bf16x8 lo, bf16x8 hi) {
  return __builtin_shufflevector(lo, hi, 0,1,2,3,4,5,6,7,8,9,10,11,12,13,14,15);
}

static __device__ __forceinline__ f32x8 wmma_bf16(bf16x16 a, bf16x16 b, f32x8 c) {
  // (neg_a, A, neg_b, B, c_mod, C, reuse_a, reuse_b)
  return __builtin_amdgcn_wmma_f32_16x16x32_bf16(false, a, false, b, (short)0, c,
                                                 false, false);
}

// Problem constants
#define BATCH 2
#define SEQ   2048
#define DMODEL 1024
#define HEADS 16
#define DHEAD 64
#define ROWS  (BATCH * SEQ)   // 4096

// ---------------------------------------------------------------------------
// Kernel 1: QKV projection.  Y[b,h,s,d] = (x @ W)[row, h*64+d] * outScale
// Block: 256 thr (8 waves), 64x64 output tile, K-step 32, bf16 via LDS.
// grid = (1024/64, 4096/64)
// ---------------------------------------------------------------------------
__global__ __launch_bounds__(256)
void qkv_proj(const float* __restrict__ X, const float* __restrict__ W,
              __bf16* __restrict__ Y, float outScale) {
  __shared__ __bf16 Xs[64][32];   // [m][k]
  __shared__ __bf16 Ws[64][32];   // [n][k] (transposed weight tile)

  const int tid  = threadIdx.x;
  const int lane = tid & 31;
  const int wv   = tid >> 5;          // wave 0..7
  const int half = lane >> 4;         // 0/1
  const int l15  = lane & 15;

  const int rowBlk = blockIdx.y * 64; // base row in [0,4096)
  const int colBlk = blockIdx.x * 64; // base col in [0,1024)
  const int mt = (wv & 3) * 16;       // wave's 16-row sub-tile
  const int nt = (wv >> 2) * 32;      // wave's 32-col sub-range (2 tiles)

  f32x8 acc0 = {}; f32x8 acc1 = {};

  for (int k0 = 0; k0 < DMODEL; k0 += 32) {
    // ---- prefetch next K-step tiles (global_prefetch_b8, one/cacheline) ----
    if (k0 + 32 < DMODEL) {
      __builtin_prefetch(X + (size_t)(rowBlk + (tid >> 2)) * DMODEL +
                             (k0 + 32) + (tid & 3) * 8, 0, 3);
      __builtin_prefetch(W + (size_t)(k0 + 32 + (tid >> 3)) * DMODEL +
                             colBlk + (tid & 7) * 8, 0, 3);
    }
    // ---- stage X (fp32 -> bf16): 64x32 tile, 8 floats / thread ----
    {
      const int r = tid >> 2;            // 0..63
      const int c = (tid & 3) * 8;       // 0,8,16,24
      const float* src = X + (size_t)(rowBlk + r) * DMODEL + k0 + c;
      float4 a = *(const float4*)(src);
      float4 b = *(const float4*)(src + 4);
      __bf16* dst = &Xs[r][c];
      dst[0] = (__bf16)a.x; dst[1] = (__bf16)a.y;
      dst[2] = (__bf16)a.z; dst[3] = (__bf16)a.w;
      dst[4] = (__bf16)b.x; dst[5] = (__bf16)b.y;
      dst[6] = (__bf16)b.z; dst[7] = (__bf16)b.w;
    }
    // ---- stage W transposed (fp32 -> bf16): 32x64 -> Ws[n][k] ----
    {
      const int kr = tid >> 3;           // 0..31
      const int cb = (tid & 7) * 8;      // 0..56
      const float* src = W + (size_t)(k0 + kr) * DMODEL + colBlk + cb;
      float4 a = *(const float4*)(src);
      float4 b = *(const float4*)(src + 4);
      Ws[cb + 0][kr] = (__bf16)a.x; Ws[cb + 1][kr] = (__bf16)a.y;
      Ws[cb + 2][kr] = (__bf16)a.z; Ws[cb + 3][kr] = (__bf16)a.w;
      Ws[cb + 4][kr] = (__bf16)b.x; Ws[cb + 5][kr] = (__bf16)b.y;
      Ws[cb + 6][kr] = (__bf16)b.z; Ws[cb + 7][kr] = (__bf16)b.w;
    }
    __syncthreads();

    // ---- fragments per ISA 16-bit A/B layout: lane half selects K range ----
    bf16x16 afrag = cat8(*(const bf16x8*)&Xs[mt + l15][8 * half],
                         *(const bf16x8*)&Xs[mt + l15][16 + 8 * half]);
    bf16x16 b0 = cat8(*(const bf16x8*)&Ws[nt + l15][8 * half],
                      *(const bf16x8*)&Ws[nt + l15][16 + 8 * half]);
    bf16x16 b1 = cat8(*(const bf16x8*)&Ws[nt + 16 + l15][8 * half],
                      *(const bf16x8*)&Ws[nt + 16 + l15][16 + 8 * half]);
    acc0 = wmma_bf16(afrag, b0, acc0);
    acc1 = wmma_bf16(afrag, b1, acc1);
    __syncthreads();
  }

  // ---- store as bf16 in [B,H,S,64] layout; D layout: row = r + 8*half ----
#pragma unroll
  for (int r = 0; r < 8; ++r) {
    const int R = rowBlk + mt + r + 8 * half;
    const int bb = R >> 11;            // /2048
    const int s  = R & 2047;
    int C = colBlk + nt + l15;
    Y[((size_t)(bb * HEADS + (C >> 6)) * SEQ + s) * DHEAD + (C & 63)] =
        (__bf16)(acc0[r] * outScale);
    C += 16;
    Y[((size_t)(bb * HEADS + (C >> 6)) * SEQ + s) * DHEAD + (C & 63)] =
        (__bf16)(acc1[r] * outScale);
  }
}

// ---------------------------------------------------------------------------
// Kernel 2: flash attention per (b,h). Block: 128 thr (4 waves), 64 query
// rows per block (16/wave), keys in chunks of 32. grid = (S/64, B*H).
// Per-head fp32 output to workspace.
// ---------------------------------------------------------------------------
__global__ __launch_bounds__(128)
void attn(const __bf16* __restrict__ Qb, const __bf16* __restrict__ Kb,
          const __bf16* __restrict__ Vb, float* __restrict__ Ohead) {
  __shared__ __bf16 Vt[64][32];       // [d][key]  (transposed V chunk)
  __shared__ __bf16 Pt[4][16][32];    // per-wave P tile (row-major)

  const int tid  = threadIdx.x;
  const int lane = tid & 31;
  const int wv   = tid >> 5;          // 0..3
  const int half = lane >> 4;
  const int l15  = lane & 15;

  const int bh   = blockIdx.y;        // 0..31
  const int qblk = blockIdx.x * 64;
  const size_t headOff = (size_t)bh * SEQ * DHEAD;
  const __bf16* Q = Qb + headOff;
  const __bf16* K = Kb + headOff;
  const __bf16* V = Vb + headOff;

  // ---- preload Q A-fragments (two K-steps over d=64); Q already has 1/T ----
  const int qbase = qblk + wv * 16;
  const __bf16* qrow = Q + (size_t)(qbase + l15) * DHEAD;
  bf16x16 qa0 = cat8(*(const bf16x8*)(qrow + 8 * half),
                     *(const bf16x8*)(qrow + 16 + 8 * half));
  bf16x16 qa1 = cat8(*(const bf16x8*)(qrow + 32 + 8 * half),
                     *(const bf16x8*)(qrow + 48 + 8 * half));

  float m[8], lsum[8];
#pragma unroll
  for (int r = 0; r < 8; ++r) { m[r] = -1e30f; lsum[r] = 0.0f; }
  f32x8 o0 = {}, o1 = {}, o2 = {}, o3 = {};

  for (int kc = 0; kc < SEQ; kc += 32) {
    // ---- prefetch next chunk's K and V rows (one cacheline per row) ----
    if (kc + 32 < SEQ) {
      __builtin_prefetch(K + (size_t)(kc + 32 + lane) * DHEAD, 0, 3);
      __builtin_prefetch(V + (size_t)(kc + 32 + (tid >> 2)) * DHEAD, 0, 3);
    }
    // ---- stage V chunk transposed into LDS: 32 keys x 64 d ----
    {
      const int key = tid >> 2;            // 0..31
      const int db  = (tid & 3) * 16;      // 0,16,32,48
      const __bf16* vrow = V + (size_t)(kc + key) * DHEAD + db;
      bf16x8 va = *(const bf16x8*)(vrow);
      bf16x8 vb = *(const bf16x8*)(vrow + 8);
#pragma unroll
      for (int i = 0; i < 8; ++i) {
        Vt[db + i][key]     = va[i];
        Vt[db + 8 + i][key] = vb[i];
      }
    }

    // ---- scores: S = Q @ K^T for two 16-key sub-tiles (K rows contiguous) --
    f32x8 s0 = {}, s1 = {};
    {
      const __bf16* k0row = K + (size_t)(kc + l15) * DHEAD;
      const __bf16* k1row = K + (size_t)(kc + 16 + l15) * DHEAD;
      bf16x16 kb;
      kb = cat8(*(const bf16x8*)(k0row + 8 * half),
                *(const bf16x8*)(k0row + 16 + 8 * half));
      s0 = wmma_bf16(qa0, kb, s0);
      kb = cat8(*(const bf16x8*)(k0row + 32 + 8 * half),
                *(const bf16x8*)(k0row + 48 + 8 * half));
      s0 = wmma_bf16(qa1, kb, s0);
      kb = cat8(*(const bf16x8*)(k1row + 8 * half),
                *(const bf16x8*)(k1row + 16 + 8 * half));
      s1 = wmma_bf16(qa0, kb, s1);
      kb = cat8(*(const bf16x8*)(k1row + 32 + 8 * half),
                *(const bf16x8*)(k1row + 48 + 8 * half));
      s1 = wmma_bf16(qa1, kb, s1);
    }

    // ---- online softmax: row = r + 8*half, cols across lanes of the half ---
    float cmax[8];
#pragma unroll
    for (int r = 0; r < 8; ++r) cmax[r] = fmaxf(s0[r], s1[r]);
#pragma unroll
    for (int mask = 1; mask <= 8; mask <<= 1)
#pragma unroll
      for (int r = 0; r < 8; ++r)
        cmax[r] = fmaxf(cmax[r], __shfl_xor(cmax[r], mask, 32));

    float p0[8], p1[8], rs[8];
#pragma unroll
    for (int r = 0; r < 8; ++r) {
      const float mn = fmaxf(m[r], cmax[r]);
      const float alpha = __expf(m[r] - mn);
      m[r] = mn;
      lsum[r] *= alpha;
      o0[r] *= alpha; o1[r] *= alpha; o2[r] *= alpha; o3[r] *= alpha;
      p0[r] = __expf(s0[r] - mn);
      p1[r] = __expf(s1[r] - mn);
      rs[r] = p0[r] + p1[r];
    }
#pragma unroll
    for (int mask = 1; mask <= 8; mask <<= 1)
#pragma unroll
      for (int r = 0; r < 8; ++r)
        rs[r] += __shfl_xor(rs[r], mask, 32);
#pragma unroll
    for (int r = 0; r < 8; ++r) lsum[r] += rs[r];

    // ---- reshape P (D layout) -> A layout via LDS ----
#pragma unroll
    for (int r = 0; r < 8; ++r) {
      const int row = r + 8 * half;
      Pt[wv][row][l15]      = (__bf16)p0[r];
      Pt[wv][row][16 + l15] = (__bf16)p1[r];
    }
    __syncthreads();   // Vt and Pt visible

    bf16x16 pa = cat8(*(const bf16x8*)&Pt[wv][l15][8 * half],
                      *(const bf16x8*)&Pt[wv][l15][16 + 8 * half]);
    // ---- O += P @ V over four d sub-tiles ----
    bf16x16 vb;
    vb = cat8(*(const bf16x8*)&Vt[0 + l15][8 * half],
              *(const bf16x8*)&Vt[0 + l15][16 + 8 * half]);
    o0 = wmma_bf16(pa, vb, o0);
    vb = cat8(*(const bf16x8*)&Vt[16 + l15][8 * half],
              *(const bf16x8*)&Vt[16 + l15][16 + 8 * half]);
    o1 = wmma_bf16(pa, vb, o1);
    vb = cat8(*(const bf16x8*)&Vt[32 + l15][8 * half],
              *(const bf16x8*)&Vt[32 + l15][16 + 8 * half]);
    o2 = wmma_bf16(pa, vb, o2);
    vb = cat8(*(const bf16x8*)&Vt[48 + l15][8 * half],
              *(const bf16x8*)&Vt[48 + l15][16 + 8 * half]);
    o3 = wmma_bf16(pa, vb, o3);
    __syncthreads();   // protect Vt/Pt before next chunk overwrite
  }

  // ---- normalize and store per-head fp32 output ----
#pragma unroll
  for (int r = 0; r < 8; ++r) {
    const float rl = 1.0f / lsum[r];
    const int q = qbase + r + 8 * half;
    float* orow = Ohead + ((size_t)bh * SEQ + q) * DHEAD;
    orow[l15]      = o0[r] * rl;
    orow[16 + l15] = o1[r] * rl;
    orow[32 + l15] = o2[r] * rl;
    orow[48 + l15] = o3[r] * rl;
  }
}

// ---------------------------------------------------------------------------
// Kernel 3: deterministic sum over heads -> d_out [B,S,64] fp32
// ---------------------------------------------------------------------------
__global__ __launch_bounds__(256)
void reduce_heads(const float* __restrict__ Oh, float* __restrict__ out) {
  const int i = blockIdx.x * blockDim.x + threadIdx.x;  // 0..262143
  const int b = i >> 17;            // / (2048*64)
  const int rem = i & 131071;
  float acc = 0.0f;
#pragma unroll
  for (int h = 0; h < HEADS; ++h)
    acc += Oh[((size_t)(b * HEADS + h) << 17) + rem];
  out[i] = acc;
}

// ---------------------------------------------------------------------------
// Launcher.  Workspace layout (bytes):
//   [0,8M)   Qb bf16 [B,H,S,64]   (pre-scaled by 1/temperature = 0.5)
//   [8,16M)  Kb bf16
//   [16,24M) Vb bf16
//   [24,40M) Ohead fp32 [B,H,S,64]
// ---------------------------------------------------------------------------
extern "C" void kernel_launch(void* const* d_in, const int* in_sizes, int n_in,
                              void* d_out, int out_size, void* d_ws, size_t ws_size,
                              hipStream_t stream) {
  const float* x  = (const float*)d_in[0];
  const float* Wk = (const float*)d_in[1];
  const float* Wq = (const float*)d_in[2];
  const float* Wv = (const float*)d_in[3];
  float* out = (float*)d_out;

  char* ws = (char*)d_ws;
  __bf16* Qb = (__bf16*)(ws);
  __bf16* Kb = (__bf16*)(ws + ((size_t)8 << 20));
  __bf16* Vb = (__bf16*)(ws + ((size_t)16 << 20));
  float*  Oh = (float*)(ws + ((size_t)24 << 20));

  dim3 gp(DMODEL / 64, ROWS / 64, 1);          // 16 x 64
  qkv_proj<<<gp, 256, 0, stream>>>(x, Wq, Qb, 0.5f);  // fold 1/T into Q
  qkv_proj<<<gp, 256, 0, stream>>>(x, Wk, Kb, 1.0f);
  qkv_proj<<<gp, 256, 0, stream>>>(x, Wv, Vb, 1.0f);

  attn<<<dim3(SEQ / 64, BATCH * HEADS), 128, 0, stream>>>(Qb, Kb, Vb, Oh);

  const int n = BATCH * SEQ * DHEAD;           // 262144
  reduce_heads<<<(n + 255) / 256, 256, 0, stream>>>(Oh, out);
}
